// ItemModel_90185723281862
// MI455X (gfx1250) — matrix operand order repaired and checked by reading, not compiled, not run on hardware
//
#include <hip/hip_runtime.h>

typedef __attribute__((ext_vector_type(16))) __bf16 v16bf;
typedef __attribute__((ext_vector_type(8)))  float  v8f;

#define EMB   64
#define BSZ   4096
#define LSEQ  200
#define LPAD  208          // 13 * 16
#define NWAVE 13
#define NTHR  (NWAVE * 32)
#define EPSV  1e-10f

union AB16 { uint4 u[2]; v16bf v; };

// ---------------------------------------------------------------------------
// Weight packer: row-major f32 W[K][N]  ->  bf16 WMMA-B fragments.
// Fragment (kt, nt) = 32 lanes x 16 bf16, lane holds column n = nt*16 + (lane&15),
// K-range kt*32 + (lane>>4)*16 .. +15 (contiguous k order used here, same as the A loads).
// ---------------------------------------------------------------------------
__global__ void pack_wmma_b(const float* __restrict__ W, int K, int N,
                            __bf16* __restrict__ out) {
    int total = (K / 32) * (N / 16) * 512;
    for (int e = blockIdx.x * blockDim.x + threadIdx.x; e < total;
         e += gridDim.x * blockDim.x) {
        int f    = e >> 9;
        int r    = e & 511;
        int lane = r >> 4;
        int t    = r & 15;
        int ntile = N >> 4;
        int kt = f / ntile, nt = f % ntile;
        int n  = nt * 16 + (lane & 15);
        int kk = kt * 32 + (lane >> 4) * 16 + t;
        out[e] = (__bf16)W[kk * N + n];
    }
}

// ---------------------------------------------------------------------------
// Per-wave GEMM: 16 rows of X (LDS bf16, row stride == K) x W (packed frags)
// -> 16x64 f32 accumulators (4 n-tiles of v8f), bias-initialized.
// ---------------------------------------------------------------------------
template<int K>
__device__ __forceinline__ void wave_gemm(const __bf16* __restrict__ X, int r0, int lane,
                                          const __bf16* __restrict__ Wf,
                                          const float* __restrict__ bias,
                                          v8f acc[4]) {
    const int m  = lane & 15;
    const int hi = lane >> 4;
#pragma unroll
    for (int nt = 0; nt < 4; ++nt) {
        float bv = bias[nt * 16 + m];
#pragma unroll
        for (int i = 0; i < 8; ++i) acc[nt][i] = bv;
    }
    const __bf16* row = X + (r0 + m) * K;
#pragma unroll
    for (int kt = 0; kt < K / 32; ++kt) {
        AB16 a;                                   // A, same k order as B frags: lane = M,
        a.u[0] = *(const uint4*)(row + kt * 32 + hi * 16);       // K  = ko..ko+7
        a.u[1] = *(const uint4*)(row + kt * 32 + hi * 16 + 8);   // K += 8
#pragma unroll
        for (int nt = 0; nt < 4; ++nt) {
            AB16 bf;
            const __bf16* bp = Wf + (size_t)((kt * 4 + nt) * 512 + lane * 16);
            bf.u[0] = *(const uint4*)(bp);
            bf.u[1] = *(const uint4*)(bp + 8);
            acc[nt] = __builtin_amdgcn_wmma_f32_16x16x32_bf16(
                false, a.v, false, bf.v, (short)0, acc[nt], false, false);
        }
    }
}

// C/D layout: VGPR i, lane -> M = i + 8*(lane>>4), N = nt*16 + (lane&15)
__device__ __forceinline__ void store_relu(__bf16* dst, int ld, int r0, int lane,
                                           v8f acc[4]) {
    const int m = lane & 15, hi = lane >> 4;
#pragma unroll
    for (int nt = 0; nt < 4; ++nt)
#pragma unroll
        for (int i = 0; i < 8; ++i) {
            float v = acc[nt][i];
            v = v > 0.f ? v : 0.f;
            dst[(r0 + i + 8 * hi) * ld + nt * 16 + m] = (__bf16)v;
        }
}

__device__ __forceinline__ void store_mul(__bf16* dst, int dld, int dcol,
                                          const __bf16* mulb, int r0, int lane,
                                          v8f acc[4]) {
    const int m = lane & 15, hi = lane >> 4;
#pragma unroll
    for (int nt = 0; nt < 4; ++nt)
#pragma unroll
        for (int i = 0; i < 8; ++i) {
            int row = r0 + i + 8 * hi, n = nt * 16 + m;
            float v = acc[nt][i] * (float)mulb[row * EMB + n];
            dst[row * dld + dcol + n] = (__bf16)v;
        }
}

__device__ __forceinline__ void store_plain(__bf16* dst, int dld, int dcol,
                                            int r0, int lane, v8f acc[4]) {
    const int m = lane & 15, hi = lane >> 4;
#pragma unroll
    for (int nt = 0; nt < 4; ++nt)
#pragma unroll
        for (int i = 0; i < 8; ++i) {
            int row = r0 + i + 8 * hi, n = nt * 16 + m;
            dst[row * dld + dcol + n] = (__bf16)acc[nt][i];
        }
}

// ---------------------------------------------------------------------------
// Attention tail: mu = exp(H . at_w2 + b)*mask, normalize, z = sum mu*F,
// zj = relu(z @ ag_w + ag_b).  F lives in left half of Abuf (stride 128).
// ---------------------------------------------------------------------------
__device__ void attn_tail(const __bf16* Hb, const __bf16* Fb,
                          const float* __restrict__ at_w2, float at_b2v,
                          const float* __restrict__ ag_w, const float* __restrict__ ag_b,
                          const float* maskv, float* mu, float* red,
                          float* zvec, float* zj, int tid) {
    if (tid < LPAD) {
        float s = at_b2v;
#pragma unroll 8
        for (int n = 0; n < EMB; ++n) s += (float)Hb[tid * EMB + n] * at_w2[n];
        mu[tid] = __expf(s) * maskv[tid];
    }
    __syncthreads();
    if (tid < 64) {
        float s = 0.f;
        for (int l = tid; l < LPAD; l += 64) s += mu[l];
        red[tid] = s;
    }
    __syncthreads();
    float denom = 0.f;
#pragma unroll 8
    for (int i = 0; i < 64; ++i) denom += red[i];
    float inv = 1.f / (denom + EPSV);
    if (tid < LPAD) mu[tid] *= inv;
    __syncthreads();
    if (tid < EMB) {
        float s = 0.f;
        for (int l = 0; l < LPAD; ++l) s += mu[l] * (float)Fb[l * 128 + tid];
        zvec[tid] = s;
    }
    __syncthreads();
    if (tid < EMB) {
        float s = ag_b[tid];
#pragma unroll 8
        for (int d = 0; d < EMB; ++d) s += zvec[d] * ag_w[d * EMB + tid];
        zj[tid] = s > 0.f ? s : 0.f;
    }
    __syncthreads();
}

// ---------------------------------------------------------------------------
// Main kernel: one block per batch item b.
// ---------------------------------------------------------------------------
__global__ __launch_bounds__(NTHR)
void item_model_kernel(const float* __restrict__ user_tab,
                       const float* __restrict__ item_tab,
                       const float* __restrict__ rating_tab,
                       const float* __restrict__ ranking_tab,
                       const float* __restrict__ species_tab,
                       const float* __restrict__ gu_b1, const float* __restrict__ gu_b2,
                       const float* __restrict__ at_b1, const float* __restrict__ at_w2,
                       const float* __restrict__ at_b2,
                       const float* __restrict__ ag_w, const float* __restrict__ ag_b,
                       const float* __restrict__ m_w1, const float* __restrict__ m_b1,
                       const float* __restrict__ m_w2, const float* __restrict__ m_b2,
                       const float* __restrict__ m_w3, const float* __restrict__ m_b3,
                       const int* __restrict__ iids,
                       const int* __restrict__ i_user_pad,
                       const int* __restrict__ i_userk_pad,
                       const int* __restrict__ i_user_species_pad,
                       const __bf16* __restrict__ guw1f,
                       const __bf16* __restrict__ guw2f,
                       const __bf16* __restrict__ atw1f,
                       float* __restrict__ out) {
    __shared__ __bf16 Xbuf[LPAD * 128];   // [p_t | spec/er/F2a]
    __shared__ __bf16 Abuf[LPAD * 128];   // [F | q]
    __shared__ __bf16 Hbuf[LPAD * EMB];
    __shared__ __bf16 Erb [LPAD * EMB];
    __shared__ __bf16 Nkb [LPAD * EMB];
    __shared__ int   s_uid[LPAD], s_rid[LPAD], s_rk[LPAD], s_sp[LPAD];
    __shared__ float maskv[LPAD], mu[LPAD], red[64], zvec[EMB];
    __shared__ float zj1[EMB], zj2[EMB], h0[2 * EMB], t1[EMB], t2[EMB];

    const int b    = blockIdx.x;
    const int tid  = threadIdx.x;
    const int lane = tid & 31;
    const int r0   = (tid >> 5) * 16;

    // ---- stage indices + mask -------------------------------------------
    for (int l = tid; l < LPAD; l += NTHR) {
        if (l < LSEQ) {
            int uid = i_user_pad[(b * LSEQ + l) * 2 + 0];
            s_uid[l] = uid;
            s_rid[l] = i_user_pad[(b * LSEQ + l) * 2 + 1];
            s_rk[l]  = i_userk_pad[(b * LSEQ + l) * 2 + 1];
            s_sp[l]  = i_user_species_pad[(b * LSEQ + l) * 2 + 1];
            maskv[l] = (uid > 0) ? 1.f : 0.f;
        } else {
            s_uid[l] = s_rid[l] = s_rk[l] = s_sp[l] = 0;
            maskv[l] = 0.f;
        }
    }
    __syncthreads();

    // ---- gather embeddings (L2-resident tables) -> LDS bf16 -------------
    const int iid = iids[b];
    for (int e = tid; e < LPAD * EMB; e += NTHR) {
        int l = e >> 6, d = e & 63;
        if (l < LSEQ) {
            Xbuf[l * 128 + d]      = (__bf16)user_tab[s_uid[l] * EMB + d];
            Xbuf[l * 128 + 64 + d] = (__bf16)species_tab[s_sp[l] * EMB + d];
            Erb[e] = (__bf16)rating_tab[s_rid[l] * EMB + d];
            Nkb[e] = (__bf16)ranking_tab[s_rk[l] * EMB + d];
        } else {
            Xbuf[l * 128 + d]      = (__bf16)0.f;
            Xbuf[l * 128 + 64 + d] = (__bf16)0.f;
            Erb[e] = (__bf16)0.f;
            Nkb[e] = (__bf16)0.f;
        }
        Abuf[l * 128 + 64 + d] = (__bf16)item_tab[iid * EMB + d];  // q broadcast
    }
    __syncthreads();

    v8f acc[4];

    // ================= branch 1: f1 = MLP2([p|spec]) * er ================
    wave_gemm<128>(Xbuf, r0, lane, guw1f, gu_b1, acc);     // relu layer
    store_relu(Hbuf, EMB, r0, lane, acc);
    __syncthreads();
    wave_gemm<64>(Hbuf, r0, lane, guw2f, gu_b2, acc);      // * er -> F1
    store_mul(Abuf, 128, 0, Erb, r0, lane, acc);
    __syncthreads();
    wave_gemm<128>(Abuf, r0, lane, atw1f, at_b1, acc);     // attention layer 1
    store_relu(Hbuf, EMB, r0, lane, acc);
    __syncthreads();
    attn_tail(Hbuf, Abuf, at_w2, at_b2[0], ag_w, ag_b, maskv, mu, red, zvec, zj1, tid);

    // ================= branch 2 ==========================================
    for (int e = tid; e < LPAD * EMB; e += NTHR)           // X2 = [p | er]
        Xbuf[(e >> 6) * 128 + 64 + (e & 63)] = Erb[e];
    __syncthreads();
    wave_gemm<128>(Xbuf, r0, lane, guw1f, gu_b1, acc);
    store_relu(Hbuf, EMB, r0, lane, acc);
    __syncthreads();
    wave_gemm<64>(Hbuf, r0, lane, guw2f, gu_b2, acc);      // * nk -> F2a -> X3 right
    store_mul(Xbuf, 128, 64, Nkb, r0, lane, acc);
    __syncthreads();
    wave_gemm<128>(Xbuf, r0, lane, guw1f, gu_b1, acc);     // X3 = [p | F2a]
    store_relu(Hbuf, EMB, r0, lane, acc);
    __syncthreads();
    wave_gemm<64>(Hbuf, r0, lane, guw2f, gu_b2, acc);      // F2 -> Abuf left
    store_plain(Abuf, 128, 0, r0, lane, acc);
    __syncthreads();
    wave_gemm<128>(Abuf, r0, lane, atw1f, at_b1, acc);     // attention layer 1
    store_relu(Hbuf, EMB, r0, lane, acc);
    __syncthreads();
    attn_tail(Hbuf, Abuf, at_w2, at_b2[0], ag_w, ag_b, maskv, mu, red, zvec, zj2, tid);

    // ================= final head ========================================
    if (tid < EMB) { h0[tid] = zj1[tid]; h0[EMB + tid] = zj2[tid]; }
    __syncthreads();
    if (tid < EMB) {
        float s = m_b1[tid];
        for (int k = 0; k < 2 * EMB; ++k) s += h0[k] * m_w1[k * EMB + tid];
        t1[tid] = s > 0.f ? s : 0.f;
    }
    __syncthreads();
    if (tid < EMB) {
        float s = m_b2[tid];
        for (int k = 0; k < EMB; ++k) s += t1[k] * m_w2[k * EMB + tid];
        t2[tid] = s > 0.f ? s : 0.f;
    }
    __syncthreads();
    if (tid < EMB) {
        float s = m_b3[tid];
        for (int k = 0; k < EMB; ++k) s += t2[k] * m_w3[k * EMB + tid];
        out[b * EMB + tid] = s > 0.f ? s : 0.f;
    }
}

// ---------------------------------------------------------------------------
extern "C" void kernel_launch(void* const* d_in, const int* in_sizes, int n_in,
                              void* d_out, int out_size, void* d_ws, size_t ws_size,
                              hipStream_t stream) {
    (void)in_sizes; (void)n_in; (void)out_size; (void)ws_size;
    const float* user_tab    = (const float*)d_in[0];
    const float* item_tab    = (const float*)d_in[1];
    const float* rating_tab  = (const float*)d_in[2];
    const float* ranking_tab = (const float*)d_in[3];
    const float* species_tab = (const float*)d_in[4];
    const float* gu_w1 = (const float*)d_in[5];
    const float* gu_b1 = (const float*)d_in[6];
    const float* gu_w2 = (const float*)d_in[7];
    const float* gu_b2 = (const float*)d_in[8];
    const float* at_w1 = (const float*)d_in[9];
    const float* at_b1 = (const float*)d_in[10];
    const float* at_w2 = (const float*)d_in[11];
    const float* at_b2 = (const float*)d_in[12];
    const float* ag_w  = (const float*)d_in[13];
    const float* ag_b  = (const float*)d_in[14];
    const float* m_w1  = (const float*)d_in[15];
    const float* m_b1  = (const float*)d_in[16];
    const float* m_w2  = (const float*)d_in[17];
    const float* m_b2  = (const float*)d_in[18];
    const float* m_w3  = (const float*)d_in[19];
    const float* m_b3  = (const float*)d_in[20];
    const int* iids               = (const int*)d_in[21];
    const int* i_user_pad         = (const int*)d_in[22];
    const int* i_userk_pad        = (const int*)d_in[23];
    const int* i_user_species_pad = (const int*)d_in[24];

    __bf16* wsb   = (__bf16*)d_ws;
    __bf16* guw1f = wsb;                 // 16 frags * 512 bf16
    __bf16* guw2f = wsb + 16 * 512;      //  8 frags
    __bf16* atw1f = wsb + 24 * 512;      // 16 frags      (total 40 KB)

    pack_wmma_b<<<40, 256, 0, stream>>>(gu_w1, 128, 64, guw1f);
    pack_wmma_b<<<40, 256, 0, stream>>>(gu_w2,  64, 64, guw2f);
    pack_wmma_b<<<40, 256, 0, stream>>>(at_w1, 128, 64, atw1f);

    item_model_kernel<<<BSZ, NTHR, 0, stream>>>(
        user_tab, item_tab, rating_tab, ranking_tab, species_tab,
        gu_b1, gu_b2, at_b1, at_w2, at_b2, ag_w, ag_b,
        m_w1, m_b1, m_w2, m_b2, m_w3, m_b3,
        iids, i_user_pad, i_userk_pad, i_user_species_pad,
        guw1f, guw2f, atw1f, (float*)d_out);
}
